// SelfAttention_21612275434249
// MI455X (gfx1250) — matrix operand order repaired and checked by reading, not compiled, not run on hardware
//
#include <hip/hip_runtime.h>
#include <hip/hip_bf16.h>
#include <stdint.h>

typedef __attribute__((ext_vector_type(16))) __bf16 v16bf;
typedef __attribute__((ext_vector_type(8)))  float  v8f;
typedef __attribute__((address_space(3))) char lds_char;

#define B_DIM 4
#define S_DIM 2048
#define D_DIM 1024

// LDS A-tile geometry: 64 rows x 64 K (bf16), row pitch padded to 144B for
// conflict-free b128 reads (36*r mod 64 distinct for r=0..15).
#define ROWP_BYTES 144
#define BUF_BYTES  (64 * ROWP_BYTES)   // 9216 B per buffer, 2 buffers

__device__ __forceinline__ unsigned short f32_to_bf16_bits(float f) {
  union { float f; unsigned u; } x; x.f = f;
  unsigned r = x.u + 0x7fffu + ((x.u >> 16) & 1u);   // round-to-nearest-even
  return (unsigned short)(r >> 16);
}
__device__ __forceinline__ float bf16_bits_to_f32(unsigned short s) {
  union { unsigned u; float f; } x; x.u = ((unsigned)s) << 16;
  return x.f;
}

// 16-byte async DMA global -> LDS (ASYNCcnt-tracked, no VGPR round trip)
__device__ __forceinline__ void async_g2l_b128(unsigned lds_byte_off,
                                               const unsigned short* gaddr) {
  asm volatile("global_load_async_to_lds_b128 %0, %1, off"
               :: "v"(lds_byte_off), "v"(gaddr) : "memory");
}
__device__ __forceinline__ void wait_async0() {
  asm volatile("s_wait_asynccnt 0x0" ::: "memory");
}

// ---------------- f32 -> bf16 convert (vectorized x4) ----------------
__global__ void k_f32_to_bf16_v4(const float4* __restrict__ in,
                                 ushort4* __restrict__ out, long long n4) {
  long long i = (long long)blockIdx.x * blockDim.x + threadIdx.x;
  if (i >= n4) return;
  float4 f = in[i];
  ushort4 o;
  o.x = f32_to_bf16_bits(f.x); o.y = f32_to_bf16_bits(f.y);
  o.z = f32_to_bf16_bits(f.z); o.w = f32_to_bf16_bits(f.w);
  out[i] = o;
}

// ---------------- V [B][S][D] -> Vt [B][D][S] via 32x32 LDS tile ----------------
__global__ void k_transpose_bf16(const unsigned short* __restrict__ in,
                                 unsigned short* __restrict__ out) {
  __shared__ unsigned short tile[32][33];
  const int tx = threadIdx.x, ty = threadIdx.y;
  const int b = blockIdx.z;
  const int d0 = blockIdx.x * 32, s0 = blockIdx.y * 32;
#pragma unroll
  for (int i = 0; i < 4; ++i) {
    int s = s0 + ty + i * 8;
    tile[ty + i * 8][tx] = in[((long long)b * S_DIM + s) * D_DIM + d0 + tx];
  }
  __syncthreads();
#pragma unroll
  for (int i = 0; i < 4; ++i) {
    int d = d0 + ty + i * 8;
    out[((long long)b * D_DIM + d) * S_DIM + s0 + tx] = tile[tx][ty + i * 8];
  }
}

// ---------------- bf16 GEMM, C = alpha*(A @ Bt^T + bias) ----------------
// A: [M][K] bf16 row-major. Bt: [N][K] bf16 row-major.
// Block = 8 waves -> 64x256 tile (waves arranged 2(M) x 4(N), 32x64 each).
// A tile (64 rows x 64 K) is double-buffered in LDS via async DMA; B direct.
__global__ void __launch_bounds__(256)
k_gemm_bt(const unsigned short* __restrict__ A,
          const unsigned short* __restrict__ Bt,
          const float* __restrict__ bias,
          unsigned short* __restrict__ outB,  // bf16 out (or null)
          float* __restrict__ outF,           // f32 out (or null)
          int M, int N, int K,
          long long strideA, long long strideB, long long strideC,
          float alpha) {
  __shared__ __align__(256) unsigned char Asmem[2][BUF_BYTES];

  const int t = threadIdx.x;
  const int lane = t & 31, wave = t >> 5;
  const int half = lane >> 4;   // 0 or 1
  const int l15  = lane & 15;
  const int mw = wave >> 2;     // 0..1 : 32-row stripe within block
  const int nw = wave & 3;      // 0..3 : 64-col stripe within block

  const int nBlk = N >> 8;                         // 256-col block tiles
  const int bm = blockIdx.x / nBlk;
  const int bn = blockIdx.x % nBlk;
  const int z = blockIdx.z;

  const unsigned short* Az = A  + (long long)z * strideA;
  const unsigned short* Bz = Bt + (long long)z * strideB;

  // --- staging assignment: thread copies 32B of the 64x64 A tile per stage ---
  const int sr   = t >> 2;        // 0..63 : A tile row
  const int sseg = (t & 3) * 2;   // 0,2,4,6 : pair of 16B segments in the row
  const unsigned short* ag = Az + (long long)(bm * 64 + sr) * K + sseg * 8;
  const unsigned ldsBase =
      (unsigned)(size_t)(lds_char*)(void*)&Asmem[0][0];
  const unsigned myOff = (unsigned)(sr * ROWP_BYTES + sseg * 16);

  // --- B fragment pointers: lane = col l15, K chunk 16*half..+15 contiguous ---
  const unsigned short* brow[4];
#pragma unroll
  for (int ni = 0; ni < 4; ++ni)
    brow[ni] = Bz + (long long)(bn * 256 + nw * 64 + ni * 16 + l15) * K + half * 16;

  union Frag { v16bf v; uint4 q[2]; };
  v8f acc[2][4];
#pragma unroll
  for (int mi = 0; mi < 2; ++mi)
#pragma unroll
    for (int ni = 0; ni < 4; ++ni)
      acc[mi][ni] = (v8f){0.f, 0.f, 0.f, 0.f, 0.f, 0.f, 0.f, 0.f};

  // prime buffer 0 with K stage 0
  async_g2l_b128(ldsBase + myOff,      ag);
  async_g2l_b128(ldsBase + myOff + 16, ag + 8);

#pragma unroll 2
  for (int k0 = 0; k0 < K; k0 += 64) {
    const int cur = (k0 >> 6) & 1;
    wait_async0();        // my DMA into buf[cur] done
    __syncthreads();      // everyone's DMA done; everyone done reading buf[cur^1]
    if (k0 + 64 < K) {    // overlap next stage's DMA with this stage's math
      const unsigned nxt = ldsBase + (unsigned)((cur ^ 1) * BUF_BYTES) + myOff;
      const unsigned short* g = ag + (k0 + 64);
      async_g2l_b128(nxt,      g);
      async_g2l_b128(nxt + 16, g + 8);
    }
    const unsigned char* Abuf = &Asmem[cur][0];
#pragma unroll
    for (int ks = 0; ks < 2; ++ks) {              // two 32-deep substeps
      Frag a[2], b[4];
#pragma unroll
      for (int mi = 0; mi < 2; ++mi) {
        const unsigned char* ap =
            Abuf + (mw * 32 + mi * 16 + l15) * ROWP_BYTES + ks * 64 + half * 32;
        a[mi].q[0] = *(const uint4*)ap;           // ds_load_b128
        a[mi].q[1] = *(const uint4*)(ap + 16);
      }
#pragma unroll
      for (int ni = 0; ni < 4; ++ni) {
        b[ni].q[0] = *(const uint4*)(brow[ni] + k0 + ks * 32);
        b[ni].q[1] = *(const uint4*)(brow[ni] + k0 + ks * 32 + 8);
      }
#pragma unroll
      for (int mi = 0; mi < 2; ++mi)
#pragma unroll
        for (int ni = 0; ni < 4; ++ni)
          acc[mi][ni] = __builtin_amdgcn_wmma_f32_16x16x32_bf16(
              false, a[mi].v, false, b[ni].v, (short)0, acc[mi][ni], false, false);
    }
  }

#pragma unroll
  for (int ni = 0; ni < 4; ++ni) {
    const int col = bn * 256 + nw * 64 + ni * 16 + l15;
    const float bval = bias ? bias[col] : 0.0f;
#pragma unroll
    for (int mi = 0; mi < 2; ++mi) {
#pragma unroll
      for (int r = 0; r < 8; ++r) {
        const int row = bm * 64 + mw * 32 + mi * 16 + r + 8 * half;
        const float v = (acc[mi][ni][r] + bval) * alpha;
        const long long idx = (long long)z * strideC + (long long)row * N + col;
        if (outB) outB[idx] = f32_to_bf16_bits(v);
        else      outF[idx] = v;
      }
    }
  }
}

// ---------------- in-place row softmax on bf16 scores ----------------
__global__ void k_softmax_rows(unsigned short* __restrict__ P) {
  __shared__ float red[256];
  const int t = threadIdx.x;
  unsigned short* row = P + (long long)blockIdx.x * S_DIM;

  float v[8];
  float mx = -3.0e38f;
#pragma unroll
  for (int i = 0; i < 8; ++i) {
    v[i] = bf16_bits_to_f32(row[t + i * 256]);
    mx = fmaxf(mx, v[i]);
  }
  red[t] = mx; __syncthreads();
  for (int s = 128; s > 0; s >>= 1) {
    if (t < s) red[t] = fmaxf(red[t], red[t + s]);
    __syncthreads();
  }
  mx = red[0]; __syncthreads();

  float sum = 0.f;
#pragma unroll
  for (int i = 0; i < 8; ++i) { v[i] = __expf(v[i] - mx); sum += v[i]; }
  red[t] = sum; __syncthreads();
  for (int s = 128; s > 0; s >>= 1) {
    if (t < s) red[t] += red[t + s];
    __syncthreads();
  }
  const float inv = 1.0f / red[0];
#pragma unroll
  for (int i = 0; i < 8; ++i) row[t + i * 256] = f32_to_bf16_bits(v[i] * inv);
}

extern "C" void kernel_launch(void* const* d_in, const int* in_sizes, int n_in,
                              void* d_out, int out_size, void* d_ws, size_t ws_size,
                              hipStream_t stream) {
  (void)in_sizes; (void)n_in; (void)out_size; (void)ws_size;
  const float* x  = (const float*)d_in[0];
  const float* Wq = (const float*)d_in[1];
  const float* bq = (const float*)d_in[2];
  const float* Wk = (const float*)d_in[3];
  const float* bk = (const float*)d_in[4];
  const float* Wv = (const float*)d_in[5];
  const float* bv = (const float*)d_in[6];
  float* out = (float*)d_out;

  const long long nx = (long long)B_DIM * S_DIM * D_DIM;  // 8M elems
  const long long nw = (long long)D_DIM * D_DIM;          // 1M elems

  unsigned short* xb  = (unsigned short*)d_ws;
  unsigned short* wqb = xb  + nx;
  unsigned short* wkb = wqb + nw;
  unsigned short* wvb = wkb + nw;
  unsigned short* Qb  = wvb + nw;
  unsigned short* Kb  = Qb  + nx;
  unsigned short* Vb  = Kb  + nx;
  unsigned short* Vt  = Vb  + nx;
  unsigned short* Sb  = Vt  + nx;   // B*S*S bf16 (scores, then probs in place)

  const dim3 blk(256);

  // 1) converts to bf16
  k_f32_to_bf16_v4<<<(unsigned)((nx / 4 + 255) / 256), blk, 0, stream>>>(
      (const float4*)x, (ushort4*)xb, nx / 4);
  k_f32_to_bf16_v4<<<(unsigned)((nw / 4 + 255) / 256), blk, 0, stream>>>(
      (const float4*)Wq, (ushort4*)wqb, nw / 4);
  k_f32_to_bf16_v4<<<(unsigned)((nw / 4 + 255) / 256), blk, 0, stream>>>(
      (const float4*)Wk, (ushort4*)wkb, nw / 4);
  k_f32_to_bf16_v4<<<(unsigned)((nw / 4 + 255) / 256), blk, 0, stream>>>(
      (const float4*)Wv, (ushort4*)wvb, nw / 4);

  // 2) QKV projections: [8192x1024] = xb @ W^T + b   (1/sqrt(D) folded into Q)
  const int M1 = B_DIM * S_DIM;                              // 8192
  const unsigned g1 = (unsigned)((M1 / 64) * (D_DIM / 256)); // 512 blocks
  k_gemm_bt<<<dim3(g1, 1, 1), blk, 0, stream>>>(xb, wqb, bq, Qb, nullptr,
      M1, D_DIM, D_DIM, 0, 0, 0, 0.03125f /* 1/sqrt(1024) */);
  k_gemm_bt<<<dim3(g1, 1, 1), blk, 0, stream>>>(xb, wkb, bk, Kb, nullptr,
      M1, D_DIM, D_DIM, 0, 0, 0, 1.0f);
  k_gemm_bt<<<dim3(g1, 1, 1), blk, 0, stream>>>(xb, wvb, bv, Vb, nullptr,
      M1, D_DIM, D_DIM, 0, 0, 0, 1.0f);

  // V transpose for the PV GEMM (makes B contiguous in K)
  k_transpose_bf16<<<dim3(D_DIM / 32, S_DIM / 32, B_DIM), dim3(32, 8), 0, stream>>>(Vb, Vt);

  // 3) scores = Qs @ K^T per batch: M=N=2048, K=1024 -> bf16
  const unsigned g2 = (unsigned)((S_DIM / 64) * (S_DIM / 256)); // 256 blocks
  k_gemm_bt<<<dim3(g2, 1, B_DIM), blk, 0, stream>>>(Qb, Kb, nullptr, Sb, nullptr,
      S_DIM, S_DIM, D_DIM,
      (long long)S_DIM * D_DIM, (long long)S_DIM * D_DIM, (long long)S_DIM * S_DIM,
      1.0f);

  // 4) row softmax in place
  k_softmax_rows<<<(unsigned)(B_DIM * S_DIM), blk, 0, stream>>>(Sb);

  // 5) out = P @ V per batch: M=2048, N=1024, K=2048 -> f32
  const unsigned g3 = (unsigned)((S_DIM / 64) * (D_DIM / 256)); // 128 blocks
  k_gemm_bt<<<dim3(g3, 1, B_DIM), blk, 0, stream>>>(Sb, Vt, nullptr, nullptr, out,
      S_DIM, D_DIM, S_DIM,
      (long long)S_DIM * S_DIM, (long long)D_DIM * S_DIM, (long long)S_DIM * D_DIM,
      1.0f);
}
